// MultiScaleMambaEncoder_48936857371345
// MI455X (gfx1250) — matrix operand order repaired and checked by reading, not compiled, not run on hardware
//
#include <hip/hip_runtime.h>
#include <hip/hip_bf16.h>

// ---------------------------------------------------------------------------
// Types for CDNA5 WMMA (wave32)
// ---------------------------------------------------------------------------
typedef __attribute__((ext_vector_type(16))) __bf16 v16bf;
typedef __attribute__((ext_vector_type(8)))  __bf16 v8bf;
typedef __attribute__((ext_vector_type(8)))  float  v8f;

#define D_MODEL 512
#define D_INNER 1024
#define D_STATE 16
#define DT_RANK 32
#define NB      4       // batch
#define L0      256     // base sequence length

__device__ __forceinline__ __bf16 f2bf(float f) {
    unsigned u = __builtin_bit_cast(unsigned, f);
    u += 0x7FFFu + ((u >> 16) & 1u);          // round-to-nearest-even
    unsigned short h = (unsigned short)(u >> 16);
    return __builtin_bit_cast(__bf16, h);
}

__device__ __forceinline__ float softplusf(float x) {
    return (x > 20.0f) ? x : log1pf(__expf(x));
}

__device__ __forceinline__ float siluf(float x) {
    return x / (1.0f + __expf(-x));
}

// ---------------------------------------------------------------------------
// fp32 -> bf16 conversion (weights, once per launch)
// ---------------------------------------------------------------------------
__global__ void cvt_bf16_kernel(const float* __restrict__ in, __bf16* __restrict__ out, int n) {
    int i = blockIdx.x * blockDim.x + threadIdx.x;
    if (i < n) out[i] = f2bf(in[i]);
}

// ---------------------------------------------------------------------------
// WMMA GEMM: C[M,N] = act( A[M,K](bf16, row-stride lda) * W[N,K]^T (bf16) + bias[N] )
// One wave computes a 16x64 C tile: 1 A-fragment reused across 4 WMMAs / K-step.
// grid = (M/16, N/64), block = 32 (one wave).
// Optional outputs: fp32 C and/or bf16 Cbf (row-major MxN).
// ---------------------------------------------------------------------------
__global__ void wmma_gemm(const __bf16* __restrict__ A, int lda,
                          const __bf16* __restrict__ W,
                          float* __restrict__ C, __bf16* __restrict__ Cbf,
                          int M, int N, int K,
                          const float* __restrict__ bias, int act) {
    const int lane = threadIdx.x & 31;
    const int tm   = blockIdx.x;
    const int tn   = blockIdx.y;         // 64-wide output tile
    const int h16  = lane & 15;
    const int g    = lane >> 4;

    const int mrow = tm * 16 + h16;                       // A-fragment row
    const __bf16* arow  = A + (size_t)mrow * lda;
    const int     nbase = tn * 64 + h16;                  // B col for sub-tile 0
    const __bf16* wrow  = W + (size_t)nbase * K + g * 16;
    const size_t  wstep = (size_t)16 * K;                 // next 16 columns

    v8f acc0 = {}, acc1 = {}, acc2 = {}, acc3 = {};

    for (int k0 = 0; k0 < K; k0 += 32) {
        // A 16x32 bf16 fragment: K = g*16 + e
        v8bf a0 = *(const v8bf*)(arow + k0 + g * 16);
        v8bf a1 = *(const v8bf*)(arow + k0 + g * 16 + 8);
        v16bf afrag = __builtin_shufflevector(a0, a1,
                        0, 1, 2, 3, 4, 5, 6, 7, 8, 9, 10, 11, 12, 13, 14, 15);
        // B 32x16 bf16 fragments: lane = col n, K = g*16 + e -> contiguous 32B
        v16bf b0 = *(const v16bf*)(wrow + k0);
        v16bf b1 = *(const v16bf*)(wrow + wstep + k0);
        v16bf b2 = *(const v16bf*)(wrow + 2 * wstep + k0);
        v16bf b3 = *(const v16bf*)(wrow + 3 * wstep + k0);

        acc0 = __builtin_amdgcn_wmma_f32_16x16x32_bf16(false, afrag, false, b0, (short)0, acc0, false, false);
        acc1 = __builtin_amdgcn_wmma_f32_16x16x32_bf16(false, afrag, false, b1, (short)0, acc1, false, false);
        acc2 = __builtin_amdgcn_wmma_f32_16x16x32_bf16(false, afrag, false, b2, (short)0, acc2, false, false);
        acc3 = __builtin_amdgcn_wmma_f32_16x16x32_bf16(false, afrag, false, b3, (short)0, acc3, false, false);
    }

    const int mbase = tm * 16 + g * 8;
#pragma unroll
    for (int j = 0; j < 4; ++j) {
        v8f accj = (j == 0) ? acc0 : (j == 1) ? acc1 : (j == 2) ? acc2 : acc3;
        const int ncol = tn * 64 + j * 16 + h16;
        const float bv = bias ? bias[ncol] : 0.0f;
#pragma unroll
        for (int r = 0; r < 8; ++r) {
            float v = accj[r] + bv;
            if (act == 1) v = softplusf(v);
            size_t off = (size_t)(mbase + r) * N + ncol;
            if (C)   C[off]   = v;
            if (Cbf) Cbf[off] = f2bf(v);
        }
    }
}

// ---------------------------------------------------------------------------
// Causal depthwise conv (4 taps) + bias + SiLU. Writes fp32 (scan) + bf16 (GEMM A).
// xz: (NB*Lc, 2*D_INNER), x part = first D_INNER cols.
// ---------------------------------------------------------------------------
__global__ void conv_silu(const float* __restrict__ xz, const float* __restrict__ cw,
                          const float* __restrict__ cb, float* __restrict__ xact,
                          __bf16* __restrict__ xact_bf, int Lc) {
    int idx = blockIdx.x * blockDim.x + threadIdx.x;
    int total = NB * Lc * D_INNER;
    if (idx >= total) return;
    int d = idx & (D_INNER - 1);
    int t = idx >> 10;               // token index b*Lc + l
    int l = t % Lc;
    float acc = cb[d];
    const float* w = cw + d * 4;
#pragma unroll
    for (int k = 0; k < 4; ++k) {
        int lp = l - 3 + k;
        if (lp >= 0)
            acc += xz[(size_t)(t - l + lp) * (2 * D_INNER) + d] * w[k];
    }
    float v = siluf(acc);
    xact[idx]    = v;
    xact_bf[idx] = f2bf(v);
}

// ---------------------------------------------------------------------------
// Selective scan: one thread per (b, d) chain, 16 states in registers.
// y = (scan + x*D) * silu(z); output bf16 (only consumed by out_proj GEMM).
// ---------------------------------------------------------------------------
__global__ void scan_kernel(const float* __restrict__ xact, const float* __restrict__ delta,
                            const float* __restrict__ xdbl, const float* __restrict__ xz,
                            const float* __restrict__ alog, const float* __restrict__ dprm,
                            __bf16* __restrict__ yact_bf, int Lc) {
    int tid = blockIdx.x * blockDim.x + threadIdx.x;
    if (tid >= NB * D_INNER) return;
    int d = tid & (D_INNER - 1);
    int b = tid >> 10;

    float A[D_STATE];
#pragma unroll
    for (int s = 0; s < D_STATE; ++s) A[s] = -__expf(alog[d * D_STATE + s]);
    float h[D_STATE] = {0.0f};
    float Dd = dprm[d];

    for (int l = 0; l < Lc; ++l) {
        size_t t = (size_t)b * Lc + l;
        float xv = xact[t * D_INNER + d];
        float dv = delta[t * D_INNER + d];
        const float* xd = xdbl + t * 64;
        float dx = dv * xv;
        float y = 0.0f;
#pragma unroll
        for (int s = 0; s < D_STATE; ++s) {
            h[s] = h[s] * __expf(dv * A[s]) + dx * xd[DT_RANK + s];
            y += h[s] * xd[DT_RANK + D_STATE + s];
        }
        float zv = xz[t * (2 * D_INNER) + D_INNER + d];
        y += xv * Dd;
        yact_bf[t * D_INNER + d] = f2bf(y * siluf(zv));
    }
}

// ---------------------------------------------------------------------------
// Layer-0 glue: cat = [s ; reverse(s)]  s:(4,256,512) -> cat_bf:(4,512,512) bf16
// ---------------------------------------------------------------------------
__global__ void build_cat(const float* __restrict__ s, __bf16* __restrict__ cat_bf) {
    int idx = blockIdx.x * blockDim.x + threadIdx.x;
    if (idx >= NB * 512 * D_MODEL) return;
    int c = idx & 511;
    int l = (idx >> 9) & 511;
    int b = idx >> 18;
    int lp = (l < 256) ? l : (511 - l);
    cat_bf[idx] = f2bf(s[((size_t)b * 256 + lp) * D_MODEL + c]);
}

// ---------------------------------------------------------------------------
// fold (0.5*(fwd + rev)) + LayerNorm + residual.  mo:(4,512,512) s/out:(4,256,512)
// one 256-thread block per output row (b,l); warpSize == 32
// ---------------------------------------------------------------------------
__global__ void fold_ln_res(const float* __restrict__ mo, const float* __restrict__ s,
                            const float* __restrict__ lnw, const float* __restrict__ lnb,
                            float* __restrict__ out) {
    int l = blockIdx.x;          // 0..255
    int b = blockIdx.y;          // 0..3
    int t = threadIdx.x;         // 0..255
    size_t rowF = ((size_t)b * 512 + l) * D_MODEL;
    size_t rowR = ((size_t)b * 512 + (511 - l)) * D_MODEL;

    float v0 = 0.5f * (mo[rowF + t]       + mo[rowR + t]);
    float v1 = 0.5f * (mo[rowF + t + 256] + mo[rowR + t + 256]);
    float ps = v0 + v1;
    float pq = v0 * v0 + v1 * v1;
    for (int o = 16; o > 0; o >>= 1) {
        ps += __shfl_down(ps, o);
        pq += __shfl_down(pq, o);
    }
    __shared__ float sh[16];
    int wv = t >> 5, ln = t & 31;
    if (ln == 0) { sh[wv * 2] = ps; sh[wv * 2 + 1] = pq; }
    __syncthreads();
    if (t == 0) {
        float S = 0.0f, Q = 0.0f;
        for (int i = 0; i < 8; ++i) { S += sh[i * 2]; Q += sh[i * 2 + 1]; }
        sh[0] = S; sh[1] = Q;
    }
    __syncthreads();
    float mean = sh[0] * (1.0f / 512.0f);
    float var  = sh[1] * (1.0f / 512.0f) - mean * mean;
    float inv  = rsqrtf(var + 1e-5f);
    size_t srow = ((size_t)b * 256 + l) * D_MODEL;
    out[srow + t]       = (v0 - mean) * inv * lnw[t]       + lnb[t]       + s[srow + t];
    out[srow + t + 256] = (v1 - mean) * inv * lnw[t + 256] + lnb[t + 256] + s[srow + t + 256];
}

// ---------------------------------------------------------------------------
// Layer-1 glue: xf = concat(a,b,c,d) along L; xfr = reverse(xf). Both bf16.
// ---------------------------------------------------------------------------
__global__ void build_xf(const float* __restrict__ sA, const float* __restrict__ sB,
                         const float* __restrict__ sC, const float* __restrict__ sD,
                         __bf16* __restrict__ xf_bf, __bf16* __restrict__ xfr_bf) {
    int idx = blockIdx.x * blockDim.x + threadIdx.x;
    if (idx >= NB * 1024 * D_MODEL) return;
    int c = idx & 511;
    int j = (idx >> 9) & 1023;
    int b = idx >> 19;
    int q = j >> 8, p = j & 255;
    const float* src = (q == 0) ? sA : (q == 1) ? sB : (q == 2) ? sC : sD;
    __bf16 v = f2bf(src[((size_t)b * 256 + p) * D_MODEL + c]);
    xf_bf[idx] = v;
    xfr_bf[(((size_t)b * 1024) + (1023 - j)) * D_MODEL + c] = v;
}

// fused = 0.5*(o4 + reverse(o5)); split into 4 outputs concatenated in d_out
__global__ void final_combine(const float* __restrict__ o4, const float* __restrict__ o5,
                              float* __restrict__ out) {
    int idx = blockIdx.x * blockDim.x + threadIdx.x;
    if (idx >= NB * 1024 * D_MODEL) return;
    int c = idx & 511;
    int j = (idx >> 9) & 1023;
    int b = idx >> 19;
    float v = 0.5f * (o4[idx] + o5[(((size_t)b * 1024) + (1023 - j)) * D_MODEL + c]);
    int q = j >> 8, l = j & 255;
    out[(size_t)q * (NB * 256 * D_MODEL) + ((size_t)b * 256 + l) * D_MODEL + c] = v;
}

// ---------------------------------------------------------------------------
// Host-side helpers
// ---------------------------------------------------------------------------
static inline void launch_gemm(const __bf16* A, int lda, const __bf16* W,
                               float* C, __bf16* Cbf,
                               int M, int N, int K, const float* bias, int act,
                               hipStream_t st) {
    dim3 grid(M / 16, N / 64), blk(32, 1, 1);
    wmma_gemm<<<grid, blk, 0, st>>>(A, lda, W, C, Cbf, M, N, K, bias, act);
}

static void mamba_run(int m, const __bf16* u_bf, float* dest, int Lc,
                      const float* conv_w, const float* conv_b, const float* A_log,
                      const float* Dprm, const float* dt_b,
                      const __bf16* in_w_bf, const __bf16* xw_bf,
                      const __bf16* dtw_bf, const __bf16* ow_bf,
                      float* xz, float* xact, __bf16* xact_bf,
                      float* xdbl, __bf16* xdbl_bf,
                      float* delta, __bf16* yact_bf,
                      hipStream_t st) {
    const int M = NB * Lc;
    // 1. xz = u @ in_w^T  (M x 2048)
    launch_gemm(u_bf, D_MODEL, in_w_bf + (size_t)m * 2 * D_INNER * D_MODEL, xz, nullptr,
                M, 2 * D_INNER, D_MODEL, nullptr, 0, st);
    // 2. causal depthwise conv + silu
    int nconv = M * D_INNER;
    conv_silu<<<(nconv + 255) / 256, 256, 0, st>>>(
        xz, conv_w + (size_t)m * D_INNER * 4, conv_b + (size_t)m * D_INNER,
        xact, xact_bf, Lc);
    // 3. x_dbl = xact @ xw^T  (M x 64), fp32 for scan + bf16 for dt GEMM
    launch_gemm(xact_bf, D_INNER, xw_bf + (size_t)m * 64 * D_INNER, xdbl, xdbl_bf,
                M, 64, D_INNER, nullptr, 0, st);
    // 4. delta = softplus(x_dbl[:, :32] @ dtw^T + dtb)  (M x 1024)
    launch_gemm(xdbl_bf, 64, dtw_bf + (size_t)m * D_INNER * DT_RANK, delta, nullptr,
                M, D_INNER, DT_RANK, dt_b + (size_t)m * D_INNER, 1, st);
    // 5. selective scan + gating -> bf16 y
    scan_kernel<<<(NB * D_INNER) / 256, 256, 0, st>>>(
        xact, delta, xdbl, xz,
        A_log + (size_t)m * D_INNER * D_STATE, Dprm + (size_t)m * D_INNER, yact_bf, Lc);
    // 6. out = y @ ow^T  (M x 512)
    launch_gemm(yact_bf, D_INNER, ow_bf + (size_t)m * D_MODEL * D_INNER, dest, nullptr,
                M, D_MODEL, D_INNER, nullptr, 0, st);
}

extern "C" void kernel_launch(void* const* d_in, const int* in_sizes, int n_in,
                              void* d_out, int out_size, void* d_ws, size_t ws_size,
                              hipStream_t stream) {
    const float* x_in[4]     = {(const float*)d_in[0], (const float*)d_in[1],
                                (const float*)d_in[2], (const float*)d_in[3]};
    const float* in_proj_w   = (const float*)d_in[4];
    const float* conv_w      = (const float*)d_in[5];
    const float* conv_b      = (const float*)d_in[6];
    const float* x_proj_w    = (const float*)d_in[7];
    const float* dt_w        = (const float*)d_in[8];
    const float* dt_b        = (const float*)d_in[9];
    const float* A_log       = (const float*)d_in[10];
    const float* Dprm        = (const float*)d_in[11];
    const float* out_proj_w  = (const float*)d_in[12];
    const float* ln_w        = (const float*)d_in[13];
    const float* ln_b        = (const float*)d_in[14];
    float* out = (float*)d_out;

    // ---- workspace layout (256B aligned slabs) ----
    char* base = (char*)d_ws;
    size_t off = 0;
    auto alloc = [&](size_t bytes) -> char* {
        char* p = base + off;
        off += (bytes + 255) & ~(size_t)255;
        return p;
    };
    __bf16* in_w_bf = (__bf16*)alloc((size_t)6 * 2 * D_INNER * D_MODEL * 2);
    __bf16* xw_bf   = (__bf16*)alloc((size_t)6 * 64 * D_INNER * 2);
    __bf16* dtw_bf  = (__bf16*)alloc((size_t)6 * D_INNER * DT_RANK * 2);
    __bf16* ow_bf   = (__bf16*)alloc((size_t)6 * D_MODEL * D_INNER * 2);
    const size_t MMAX = (size_t)NB * 1024;                 // max tokens per call
    __bf16* cat_bf  = (__bf16*)alloc(MMAX * D_MODEL * 2);  // layer0 cat / layer1 xf
    __bf16* xfr_bf  = (__bf16*)alloc(MMAX * D_MODEL * 2);
    float*  xz      = (float*) alloc(MMAX * 2 * D_INNER * 4);
    float*  xact    = (float*) alloc(MMAX * D_INNER * 4);
    __bf16* xact_bf = (__bf16*)alloc(MMAX * D_INNER * 2);
    float*  xdbl    = (float*) alloc(MMAX * 64 * 4);
    __bf16* xdbl_bf = (__bf16*)alloc(MMAX * 64 * 2);
    float*  delta   = (float*) alloc(MMAX * D_INNER * 4);
    __bf16* yact_bf = (__bf16*)alloc(MMAX * D_INNER * 2);
    float*  mout    = (float*) alloc(MMAX * D_MODEL * 4);
    float*  sbuf[4];
    for (int q = 0; q < 4; ++q) sbuf[q] = (float*)alloc((size_t)NB * 256 * D_MODEL * 4);
    float* out4 = (float*)alloc(MMAX * D_MODEL * 4);
    float* out5 = (float*)alloc(MMAX * D_MODEL * 4);
    (void)ws_size; (void)in_sizes; (void)n_in; (void)out_size;

    // ---- weight conversion fp32 -> bf16 ----
    {
        int n;
        n = 6 * 2 * D_INNER * D_MODEL;
        cvt_bf16_kernel<<<(n + 255) / 256, 256, 0, stream>>>(in_proj_w, in_w_bf, n);
        n = 6 * 64 * D_INNER;
        cvt_bf16_kernel<<<(n + 255) / 256, 256, 0, stream>>>(x_proj_w, xw_bf, n);
        n = 6 * D_INNER * DT_RANK;
        cvt_bf16_kernel<<<(n + 255) / 256, 256, 0, stream>>>(dt_w, dtw_bf, n);
        n = 6 * D_MODEL * D_INNER;
        cvt_bf16_kernel<<<(n + 255) / 256, 256, 0, stream>>>(out_proj_w, ow_bf, n);
    }

    // ---- layer 0: four dual-direction mamba blocks ----
    // dual(0, x0_hw)->sbuf0, dual(2, x1_hw)->sbuf1, dual(1, x0_wh)->sbuf2, dual(3, x1_wh)->sbuf3
    const int mamba_idx[4] = {0, 2, 1, 3};
    for (int q = 0; q < 4; ++q) {
        int ncat = NB * 512 * D_MODEL;
        build_cat<<<(ncat + 255) / 256, 256, 0, stream>>>(x_in[q], cat_bf);
        mamba_run(mamba_idx[q], cat_bf, mout, 512,
                  conv_w, conv_b, A_log, Dprm, dt_b,
                  in_w_bf, xw_bf, dtw_bf, ow_bf,
                  xz, xact, xact_bf, xdbl, xdbl_bf, delta, yact_bf, stream);
        fold_ln_res<<<dim3(256, NB), 256, 0, stream>>>(mout, x_in[q], ln_w, ln_b, sbuf[q]);
    }

    // ---- layer 1: concat + forward/backward mamba ----
    int nxf = NB * 1024 * D_MODEL;
    build_xf<<<(nxf + 255) / 256, 256, 0, stream>>>(sbuf[0], sbuf[1], sbuf[2], sbuf[3],
                                                    cat_bf, xfr_bf);
    mamba_run(4, cat_bf, out4, 1024, conv_w, conv_b, A_log, Dprm, dt_b,
              in_w_bf, xw_bf, dtw_bf, ow_bf,
              xz, xact, xact_bf, xdbl, xdbl_bf, delta, yact_bf, stream);
    mamba_run(5, xfr_bf, out5, 1024, conv_w, conv_b, A_log, Dprm, dt_b,
              in_w_bf, xw_bf, dtw_bf, ow_bf,
              xz, xact, xact_bf, xdbl, xdbl_bf, delta, yact_bf, stream);
    final_combine<<<(nxf + 255) / 256, 256, 0, stream>>>(out4, out5, out);
}
